// GCNLayer_8589934618
// MI455X (gfx1250) — matrix operand order repaired and checked by reading, not compiled, not run on hardware
//
#include <hip/hip_runtime.h>

typedef __attribute__((ext_vector_type(16))) __bf16 v16bf;
typedef __attribute__((ext_vector_type(8)))  float  v8f;

#define D 128
#define BN_EPS 1e-5f

// ---------------- init: out = bias broadcast, deg = 0, sums/sumsq = 0 --------
__global__ void gcn_init_kernel(const float* __restrict__ bias,
                                float* __restrict__ out,
                                float* __restrict__ deg,
                                float* __restrict__ sums, // 256 floats: sums|sumsq
                                int N) {
  int i = blockIdx.x * blockDim.x + threadIdx.x;
  long long total = (long long)N * D;
  if (i < total) out[i] = bias[i & (D - 1)];
  if (i < N)     deg[i] = 0.0f;
  if (i < 256)   sums[i] = 0.0f;
}

// ---------------- degree: atomic count over dst ------------------------------
__global__ void gcn_degree_kernel(const long long* __restrict__ dst,
                                  float* __restrict__ deg, int E) {
  int e = blockIdx.x * blockDim.x + threadIdx.x;
  if (e < E) atomicAdd(&deg[(int)dst[e]], 1.0f);
}

// ---------------- dinv = deg>0 ? rsqrt(deg) : 0 ------------------------------
__global__ void gcn_dinv_kernel(const float* __restrict__ deg,
                                float* __restrict__ dinv, int N) {
  int i = blockIdx.x * blockDim.x + threadIdx.x;
  if (i >= N) return;
  float dg = deg[i];
  dinv[i] = dg > 0.0f ? rsqrtf(dg) : 0.0f;
}

// ---------------- xw = x @ W^T via v_wmma_f32_16x16x32_bf16 ------------------
// One wave per 16-row M-tile; loops 8 N-tiles x 4 K-tiles. x,W converted to
// bf16 in registers (fp32 accumulation in the WMMA).
__global__ void gcn_gemm_wmma_kernel(const float* __restrict__ x,
                                     const float* __restrict__ W,
                                     float* __restrict__ xw, int N) {
  int wave  = (blockIdx.x * blockDim.x + threadIdx.x) >> 5;
  int lane  = threadIdx.x & 31;
  int mtile = wave;
  if (mtile * 16 >= N) return;

  int lrow  = lane & 15;       // row / col within tile
  int khalf = lane >> 4;       // 0 or 1

  // A fragments: row = mtile*16 + lrow; per k-tile, VGPR0-3 hold K = kt*32 +
  // khalf*8 + 0..7, VGPR4-7 hold the same +16.
  const float* xr = x + (long long)(mtile * 16 + lrow) * D;
  v16bf a[4];
#pragma unroll
  for (int kt = 0; kt < 4; ++kt) {
    int k0 = kt * 32 + khalf * 8;
#pragma unroll
    for (int j = 0; j < 8; ++j) {
      a[kt][j]     = (__bf16)xr[k0 + j];
      a[kt][8 + j] = (__bf16)xr[k0 + 16 + j];
    }
  }

#pragma unroll
  for (int nt = 0; nt < 8; ++nt) {
    // B = W^T: lane holds col o = nt*16 + lrow, K = kt*32 + khalf*8 + 0..7
    // and the same +16, from W row o.
    const float* wr = W + (long long)(nt * 16 + lrow) * D;
    v8f acc = {};
#pragma unroll
    for (int kt = 0; kt < 4; ++kt) {
      int k0 = kt * 32 + khalf * 8;
      v16bf b;
#pragma unroll
      for (int j = 0; j < 16; ++j) b[j] = (__bf16)wr[k0 + (j & 7) + ((j >> 3) << 4)];
      acc = __builtin_amdgcn_wmma_f32_16x16x32_bf16(
          false, a[kt], false, b, (short)0, acc, false, false);
    }
    // D layout: VGPR r -> row r + 8*khalf, col lrow
    float* op = xw + ((long long)(mtile * 16 + 8 * khalf)) * D + nt * 16 + lrow;
#pragma unroll
    for (int r = 0; r < 8; ++r) op[(long long)r * D] = acc[r];
  }
}

// ---------------- edge phase: one wave per edge, float4 gather + f32 atomics -
__global__ void gcn_scatter_kernel(const long long* __restrict__ src,
                                   const long long* __restrict__ dst,
                                   const float* __restrict__ dinv,
                                   const float* __restrict__ xw,
                                   float* __restrict__ agg, int E) {
  long long tid  = (long long)blockIdx.x * blockDim.x + threadIdx.x;
  int       lane = threadIdx.x & 31;
  long long e    = tid >> 5;
  if (e >= E) return;
  int s = (int)src[e];
  int d = (int)dst[e];
  float norm = dinv[s] * dinv[d];
  const float4 v = ((const float4*)(xw + (long long)s * D))[lane];
  float* p = agg + (long long)d * D + lane * 4;
  atomicAdd(p + 0, v.x * norm);
  atomicAdd(p + 1, v.y * norm);
  atomicAdd(p + 2, v.z * norm);
  atomicAdd(p + 3, v.w * norm);
}

// ---------------- BN stats: per-feature sum / sumsq --------------------------
__global__ void gcn_bnstats_kernel(const float* __restrict__ agg,
                                   float* __restrict__ sums,
                                   float* __restrict__ sumsq, int N) {
  __shared__ float ssum[256];
  __shared__ float ssq[256];
  int tid  = threadIdx.x;
  int d    = tid & (D - 1);
  int half = tid >> 7;
  float s = 0.0f, q = 0.0f;
  for (int n = blockIdx.x * 2 + half; n < N; n += gridDim.x * 2) {
    float v = agg[(long long)n * D + d];
    s += v;
    q += v * v;
  }
  ssum[tid] = s;
  ssq[tid]  = q;
  __syncthreads();
  if (tid < 128) {
    atomicAdd(&sums[d],  ssum[tid] + ssum[tid + 128]);
    atomicAdd(&sumsq[d], ssq[tid]  + ssq[tid + 128]);
  }
}

// ---------------- finalize: BN + ReLU + residual, in place on d_out ----------
__global__ void gcn_final_kernel(const float* __restrict__ x,
                                 const float* __restrict__ sums,
                                 const float* __restrict__ sumsq,
                                 const float* __restrict__ gamma,
                                 const float* __restrict__ beta,
                                 float* __restrict__ out,
                                 long long total, float invN) {
  long long i = (long long)blockIdx.x * blockDim.x + threadIdx.x;
  if (i >= total) return;
  int d = (int)(i & (D - 1));
  float mu  = sums[d] * invN;
  float var = sumsq[d] * invN - mu * mu;
  float inv = rsqrtf(var + BN_EPS);
  float h = (out[i] - mu) * inv * gamma[d] + beta[d];
  h = h > 0.0f ? h : 0.0f;
  out[i] = h + x[i];
}

extern "C" void kernel_launch(void* const* d_in, const int* in_sizes, int n_in,
                              void* d_out, int out_size, void* d_ws, size_t ws_size,
                              hipStream_t stream) {
  const float*     x     = (const float*)d_in[0];
  const long long* eidx  = (const long long*)d_in[1];
  const float*     W     = (const float*)d_in[2];
  const float*     bias  = (const float*)d_in[3];
  const float*     gamma = (const float*)d_in[4];
  const float*     beta  = (const float*)d_in[5];
  float*           out   = (float*)d_out;

  const int N = in_sizes[0] / D;        // 100000
  const int E = in_sizes[1] / 2;        // 1600000
  const long long* src = eidx;
  const long long* dst = eidx + E;

  // workspace layout (floats): deg[N] | dinv[N] | sums[128] | sumsq[128] | xw[N*D]
  float* deg   = (float*)d_ws;
  float* dinv  = deg + N;
  float* sums  = dinv + N;
  float* sumsq = sums + 128;
  float* xw    = sums + 256;            // 2N+256 floats in -> 16B aligned

  const long long total = (long long)N * D;

  gcn_init_kernel<<<(int)((total + 255) / 256), 256, 0, stream>>>(bias, out, deg, sums, N);
  gcn_degree_kernel<<<(E + 255) / 256, 256, 0, stream>>>(dst, deg, E);
  gcn_dinv_kernel<<<(N + 255) / 256, 256, 0, stream>>>(deg, dinv, N);

  const int mtiles = (N + 15) / 16;                 // 6250 waves
  const int gemm_blocks = (mtiles + 7) / 8;         // 8 waves / 256-thread block
  gcn_gemm_wmma_kernel<<<gemm_blocks, 256, 0, stream>>>(x, W, xw, N);

  const int scat_blocks = (E + 7) / 8;              // 8 edges (waves) per block
  gcn_scatter_kernel<<<scat_blocks, 256, 0, stream>>>(src, dst, dinv, xw, out, E);

  gcn_bnstats_kernel<<<512, 256, 0, stream>>>(out, sums, sumsq, N);

  gcn_final_kernel<<<(int)((total + 255) / 256), 256, 0, stream>>>(
      x, sums, sumsq, gamma, beta, out, total, 1.0f / (float)N);
}
